// DensePoseV1ConvXGNInsHead_78237124264116
// MI455X (gfx1250) — matrix operand order repaired and checked by reading, not compiled, not run on hardware
//
#include <hip/hip_runtime.h>

typedef __attribute__((ext_vector_type(16))) __bf16 v16bf;
typedef __attribute__((ext_vector_type(8)))  float v8f;

#define CIN  64
#define HID  64
#define HH   256
#define WW   256
#define NB   8
#define KBCT 18         // 576 / 32
#define SEGS 128        // 8 images * 16 instance ids
#define PW   258        // padded spatial extent (1-pixel zero border)

__device__ __forceinline__ unsigned short f2bf(float f) {
    unsigned int u = __float_as_uint(f);
    u += 0x7FFFu + ((u >> 16) & 1u);          // round to nearest even
    return (unsigned short)(u >> 16);
}

// ---------------------------------------------------------------------------
// Pack conv weights [O=64][I=64][3][3] f32 into WMMA B-fragment order:
// dst[ntile][kb][lane][v] = bf16 pair (K, K+1) for column N = ntile*16+lane%16,
// with K = kb*32 + (lane/16)*16 + 2*v and K -> (tap, cin) as K = tap*64 + cin.
// ---------------------------------------------------------------------------
__global__ void pack_weights(const float* __restrict__ w, unsigned int* __restrict__ dst) {
    int idx = blockIdx.x * blockDim.x + threadIdx.x;
    if (idx >= 4 * KBCT * 32 * 8) return;
    int v     = idx & 7;
    int l     = (idx >> 3) & 31;
    int kb    = (idx >> 8) % KBCT;
    int ntile = idx / (8 * 32 * KBCT);
    int n     = ntile * 16 + (l & 15);
    int g     = l >> 4;
    int k     = kb * 32 + g * 16 + v * 2;
    int tap   = k >> 6;
    int c     = k & 63;
    int dh    = tap / 3, dw = tap % 3;
    float f0  = w[((n * CIN + c    ) * 3 + dh) * 3 + dw];
    float f1  = w[((n * CIN + c + 1) * 3 + dh) * 3 + dw];
    dst[idx]  = (unsigned int)f2bf(f0) | ((unsigned int)f2bf(f1) << 16);
}

// ---------------------------------------------------------------------------
// Zero the 1-pixel border of the padded bf16 NHWC activation buffer
// xb[n][PW][PW][64]. Interior is overwritten every layer; borders stay zero.
// ---------------------------------------------------------------------------
__global__ void zero_pad_borders(unsigned short* __restrict__ xb) {
    int p = blockIdx.x * 256 + threadIdx.x;        // over NB*PW*PW pixels
    if (p >= NB * PW * PW) return;
    int pix = p % (PW * PW);
    int hh = pix / PW, ww = pix % PW;
    if (hh == 0 || hh == PW - 1 || ww == 0 || ww == PW - 1) {
        uint4 z = {0u, 0u, 0u, 0u};
        uint4* d = (uint4*)(xb + (size_t)p * 64);
#pragma unroll
        for (int i = 0; i < 8; ++i) d[i] = z;
    }
}

// f32 NCHW -> bf16 padded NHWC (interior only)
__global__ void f32_to_bf16_nhwc(const float* __restrict__ x, unsigned short* __restrict__ xb) {
    int t = blockIdx.x * 256 + threadIdx.x;        // over N*C*H*W = 2^25
    int w  = t & 255;
    int hh = (t >> 8) & 255;
    int c  = (t >> 16) & 63;
    int n  = t >> 22;
    xb[(((size_t)(n * PW + hh + 1)) * PW + (w + 1)) * 64 + c] = f2bf(x[t]);
}

// ---------------------------------------------------------------------------
// Implicit-GEMM 3x3 conv via v_wmma_f32_16x16x32_bf16.
// Input: bf16 zero-padded NHWC (channel-fastest = LDS fragment layout).
// Block: 128 threads (4 waves) -> 2 output rows x 64 cols x 64 out channels.
// Halo (4 rows x 66 cols x 64ch bf16, 33 KB) staged with async-to-LDS b128.
// ---------------------------------------------------------------------------
__global__ __launch_bounds__(128) void conv3x3_wmma(
    const unsigned short* __restrict__ xb, const unsigned int* __restrict__ wpack,
    float* __restrict__ out) {
    __shared__ unsigned short sHalo[4 * 66 * 64];   // [row][col][cin] bf16

    int b   = blockIdx.x;
    int n   = b >> 9;             // 512 blocks per image (128 row-pairs * 4 col tiles)
    int rem = b & 511;
    int h0  = (rem >> 2) << 1;    // first output row of the pair
    int w0  = (rem & 3) << 6;     // 64-pixel column tile
    int tid = threadIdx.x;

    // Async halo copy: 4 padded rows (h0 .. h0+3), cols w0 .. w0+65, all 64 ch.
    // Each row chunk is contiguous: 66 px * 128 B = 8448 B. 2112 b128 units.
    {
        unsigned sbase = (unsigned)(uintptr_t)&sHalo[0];
        for (int u = tid; u < 4 * 528; u += 128) {
            int row  = u / 528;
            int rofs = (u - row * 528) * 16;     // byte offset within the row chunk
            unsigned lds = sbase + row * 8448 + rofs;
            unsigned long long ga = (unsigned long long)(uintptr_t)(
                xb + (((size_t)(n * PW + h0 + row)) * PW + w0) * 64) + (unsigned)rofs;
            asm volatile("global_load_async_to_lds_b128 %0, %1, off"
                         :: "v"(lds), "v"(ga) : "memory");
        }
        asm volatile("s_wait_asynccnt 0" ::: "memory");
    }
    __syncthreads();

    const unsigned int* sH32 = (const unsigned int*)sHalo;
    int wave = tid >> 5;
    int lane = tid & 31;
    int g    = lane >> 4;
    int lm   = lane & 15;

    v8f zero = {0.f, 0.f, 0.f, 0.f, 0.f, 0.f, 0.f, 0.f};
    v8f acc[8] = {zero, zero, zero, zero, zero, zero, zero, zero};

    const uint4* bb = (const uint4*)wpack;

    for (int kb = 0; kb < KBCT; ++kb) {
        // B fragment: two 128-bit global loads from the packed (L2-hot) table
        union { uint4 q[2]; v16bf v; } bf;
        int bidx = ((wave * KBCT + kb) * 32 + lane) * 2;
        bf.q[0] = bb[bidx];
        bf.q[1] = bb[bidx + 1];

        int tap = kb >> 1;
        int cb2 = (kb & 1) << 4;          // (c base)/2 in dwords
        int trow = tap / 3;
        int dwc  = tap - trow * 3;

#pragma unroll
        for (int m = 0; m < 8; ++m) {
            int rloc   = m >> 2;          // output row within the pair
            int mc     = m & 3;           // 16-pixel column tile
            int col    = (mc << 4) + lm + dwc;
            int base32 = ((rloc + trow) * 66 + col) * 32 + cb2;
            union { unsigned int u[8]; v16bf v; } af;
#pragma unroll
            for (int vv = 0; vv < 8; ++vv) {
                // k order chosen here (same as B pack): VGPR vv holds K = g*16 + 2*vv
                int kl = (g << 4) + (vv << 1);
                af.u[vv] = sH32[base32 + (kl >> 1)];
            }
            acc[m] = __builtin_amdgcn_wmma_f32_16x16x32_bf16(
                false, af.v, false, bf.v, (short)0, acc[m], false, false);
        }
    }

    // C/D layout: VGPR r, lanes 0-15 -> M=r, lanes 16-31 -> M=r+8; N = lane%16
    int oc   = (wave << 4) + lm;
    int poff = g ? 8 : 0;
#pragma unroll
    for (int m = 0; m < 8; ++m) {
        int rloc = m >> 2;
        int mc   = m & 3;
        float* obase = out + ((n * HID + oc) * HH + h0 + rloc) * WW + w0 + (mc << 4) + poff;
#pragma unroll
        for (int r = 0; r < 8; ++r)
            obase[r] = acc[m][r];
    }
}

// ---------------------------------------------------------------------------
// Per-instance BN helpers
// ---------------------------------------------------------------------------
__global__ void zero_stats(float* __restrict__ s, int nfloats) {
    int i = blockIdx.x * blockDim.x + threadIdx.x;
    if (i < nfloats) s[i] = 0.f;
}

__global__ void count_pixels(const int* __restrict__ ids, float* __restrict__ cnt) {
    __shared__ float lc[16];
    int tid = threadIdx.x;
    if (tid < 16) lc[tid] = 0.f;
    __syncthreads();
    int p  = blockIdx.x * 256 + tid;      // over N*H*W; one image per block
    int n  = blockIdx.x >> 8;
    int id = ids[p];
    if (id >= 0) atomicAdd(&lc[id], 1.f);
    __syncthreads();
    if (tid < 16 && lc[tid] != 0.f) atomicAdd(&cnt[n * 16 + tid], lc[tid]);
}

__global__ void accum_stats(const float* __restrict__ x, const int* __restrict__ ids,
                            float* __restrict__ sum, float* __restrict__ sq) {
    __shared__ float ls[16], lq[16];
    int tid = threadIdx.x;
    if (tid < 16) { ls[tid] = 0.f; lq[tid] = 0.f; }
    __syncthreads();
    int blk  = blockIdx.x;                // over N*C*H rows
    int hrow = blk % HH;
    int c    = (blk / HH) % HID;
    int n    = blk / (HH * HID);
    float v  = x[((n * HID + c) * HH + hrow) * WW + tid];
    int   id = ids[(n * HH + hrow) * WW + tid];
    if (id >= 0) { atomicAdd(&ls[id], v); atomicAdd(&lq[id], v * v); }
    __syncthreads();
    if (tid < 16) {
        if (ls[tid] != 0.f) atomicAdd(&sum[(n * 16 + tid) * HID + c], ls[tid]);
        if (lq[tid] != 0.f) atomicAdd(&sq[(n * 16 + tid) * HID + c], lq[tid]);
    }
}

__global__ void finalize_stats(const float* __restrict__ cnt, const float* __restrict__ sum,
                               const float* __restrict__ sq, const float* __restrict__ gamma,
                               const float* __restrict__ beta, float* __restrict__ scale,
                               float* __restrict__ shift) {
    int t = blockIdx.x * blockDim.x + threadIdx.x;
    if (t >= SEGS * HID) return;
    int seg = t / HID, c = t % HID;
    float cn  = cnt[seg]; cn = cn < 1.f ? 1.f : cn;
    float m   = sum[t] / cn;
    float var = sq[t] / cn - m * m;
    float inv = rsqrtf(var + 1e-5f);
    float sc  = gamma[c] * inv;
    scale[t]  = sc;
    shift[t]  = beta[c] - m * sc;
}

// BN + ReLU fused with f32->bf16 NHWC repack (feeds the next conv)
__global__ void norm_relu_bf16(const float* __restrict__ x, const int* __restrict__ ids,
                               const float* __restrict__ scale, const float* __restrict__ shift,
                               unsigned short* __restrict__ xb) {
    int t  = blockIdx.x * 256 + threadIdx.x;   // over N*C*H*W = 2^25
    int w  = t & 255;
    int hh = (t >> 8) & 255;
    int c  = (t >> 16) & 63;
    int n  = t >> 22;
    float v = x[t];
    int id  = ids[(n << 16) + (hh << 8) + w];
    if (id >= 0) {
        int s = (n * 16 + id) * HID + c;
        v = v * scale[s] + shift[s];
    }
    v = v > 0.f ? v : 0.f;
    xb[(((size_t)(n * PW + hh + 1)) * PW + (w + 1)) * 64 + c] = f2bf(v);
}

// Final BN + ReLU, f32 in place
__global__ void norm_relu(float* __restrict__ x, const int* __restrict__ ids,
                          const float* __restrict__ scale, const float* __restrict__ shift) {
    int t  = blockIdx.x * 256 + threadIdx.x;
    int hw = t & 65535;
    int c  = (t >> 16) & 63;
    int n  = t >> 22;
    float v = x[t];
    int id  = ids[(n << 16) + hw];
    if (id >= 0) {
        int s = (n * 16 + id) * HID + c;
        v = v * scale[s] + shift[s];
    }
    x[t] = v > 0.f ? v : 0.f;
}

// ---------------------------------------------------------------------------
extern "C" void kernel_launch(void* const* d_in, const int* in_sizes, int n_in,
                              void* d_out, int out_size, void* d_ws, size_t ws_size,
                              hipStream_t stream) {
    const float* feat = (const float*)d_in[0];
    const int*   ids  = (const int*)d_in[1];
    const float* w0   = (const float*)d_in[2];
    const float* g0   = (const float*)d_in[3];
    const float* b0   = (const float*)d_in[4];
    const float* w1   = (const float*)d_in[5];
    const float* g1   = (const float*)d_in[6];
    const float* b1   = (const float*)d_in[7];

    float* ws = (float*)d_ws;
    unsigned int* wp0 = (unsigned int*)ws;                 // 18432 dwords
    unsigned int* wp1 = wp0 + 18432;                       // 18432 dwords
    float* cnt   = ws + 36864;                             // 128
    float* sum   = cnt + SEGS;                             // 8192
    float* sq    = sum + SEGS * HID;                       // 8192
    float* scale = sq  + SEGS * HID;                       // 8192
    float* shift = scale + SEGS * HID;                     // 8192
    unsigned short* xbuf = (unsigned short*)(ws + 69760);  // NB*PW*PW*64 bf16
    float* y1 = ws + 69760 + (NB * PW * PW * 64) / 2;      // N*C*H*W f32
    float* out = (float*)d_out;

    const int packN  = 4 * KBCT * 32 * 8;                  // 18432
    const int statsN = SEGS + 2 * SEGS * HID;              // cnt+sum+sq contiguous
    const int convBlocks = NB * (HH / 2) * 4;              // 4096
    const int elemBlocks = (NB * HID * HH * WW) / 256;     // 131072
    const int padBlocks  = (NB * PW * PW + 255) / 256;

    pack_weights<<<(packN + 255) / 256, 256, 0, stream>>>(w0, wp0);
    pack_weights<<<(packN + 255) / 256, 256, 0, stream>>>(w1, wp1);
    zero_pad_borders<<<padBlocks, 256, 0, stream>>>(xbuf);

    // Layer 1: repack -> conv -> instance BN -> (ReLU + bf16 repack)
    f32_to_bf16_nhwc<<<elemBlocks, 256, 0, stream>>>(feat, xbuf);
    conv3x3_wmma<<<convBlocks, 128, 0, stream>>>(xbuf, wp0, y1);
    zero_stats<<<(statsN + 255) / 256, 256, 0, stream>>>(cnt, statsN);
    count_pixels<<<NB * HH * WW / 256, 256, 0, stream>>>(ids, cnt);
    accum_stats<<<NB * HID * HH, 256, 0, stream>>>(y1, ids, sum, sq);
    finalize_stats<<<(SEGS * HID + 127) / 128, 128, 0, stream>>>(cnt, sum, sq, g0, b0, scale, shift);
    norm_relu_bf16<<<elemBlocks, 256, 0, stream>>>(y1, ids, scale, shift, xbuf);

    // Layer 2: conv -> instance BN -> ReLU (in place on d_out)
    conv3x3_wmma<<<convBlocks, 128, 0, stream>>>(xbuf, wp1, out);
    zero_stats<<<(statsN + 255) / 256, 256, 0, stream>>>(cnt, statsN);
    count_pixels<<<NB * HH * WW / 256, 256, 0, stream>>>(ids, cnt);
    accum_stats<<<NB * HID * HH, 256, 0, stream>>>(out, ids, sum, sq);
    finalize_stats<<<(SEGS * HID + 127) / 128, 128, 0, stream>>>(cnt, sum, sq, g1, b1, scale, shift);
    norm_relu<<<elemBlocks, 256, 0, stream>>>(out, ids, scale, shift);
}
